// ESIMAttention_31447750541790
// MI455X (gfx1250) — matrix operand order repaired and checked by reading, not compiled, not run on hardware
//
#include <hip/hip_runtime.h>
#include <hip/hip_bf16.h>

// ---------------------------------------------------------------------------
// ESIM attention on MI455X (gfx1250, wave32, WMMA).
//   e  = x1 @ x2^T          [32,512,512]  (bf16 WMMA, f32 accum)
//   e1 = softmax(e, axis=2), e2 = softmax(e, axis=1)
//   xe1 = e1 @ x2, xe2 = e2 @ x1   -> d_out = [xe1 ; xe2] fp32
// e lives in d_ws (33.5MB, L2-resident). bf16 copies of x1/x2 halve GEMM
// operand bytes. Softmax weights are regenerated in-register in the second
// GEMM (v_exp_f32 -> bf16 A-fragments), row normalization folded into the
// epilogue. Prefetches are unconditional (target stays inside d_ws) so the
// GEMM1 inner loop carries no branches besides the loop back-edge.
// ---------------------------------------------------------------------------

typedef __attribute__((ext_vector_type(16))) __bf16 v16bf;
typedef __attribute__((ext_vector_type(8)))  __bf16 v8bf;
typedef __attribute__((ext_vector_type(4)))  __bf16 bf16x4;
typedef __attribute__((ext_vector_type(8)))  float  v8f;
typedef __attribute__((ext_vector_type(4)))  float  f32x4;

#define NB 32
#define LL 512
#define DD 768

// ---------------- fp32 -> bf16 convert (vectorized) ----------------
__global__ __launch_bounds__(256) void k_cvt(const float* __restrict__ x,
                                             __bf16* __restrict__ y, int n4) {
  int t = blockIdx.x * 256 + threadIdx.x;
  if (t >= n4) return;
  f32x4 v = ((const f32x4*)x)[t];
  bf16x4 o;
  o[0] = (__bf16)v[0]; o[1] = (__bf16)v[1];
  o[2] = (__bf16)v[2]; o[3] = (__bf16)v[3];
  ((bf16x4*)y)[t] = o;
}

// ---------------- GEMM1: e[b] = x1b[b] @ x2b[b]^T ----------------
// block = 256 thr = 8 waves arranged 2(i) x 4(j); wave tile 64x64.
__global__ __launch_bounds__(256) void k_gemm1(const __bf16* __restrict__ x1b,
                                               const __bf16* __restrict__ x2b,
                                               float* __restrict__ e) {
  const int lane = threadIdx.x & 31;
  const int wave = threadIdx.x >> 5;
  const int b  = blockIdx.z;
  const int i0 = blockIdx.x * 128 + (wave >> 2) * 64;
  const int j0 = blockIdx.y * 256 + (wave & 3) * 64;
  const __bf16* A  = x1b + (size_t)b * LL * DD;
  const __bf16* Bm = x2b + (size_t)b * LL * DD;
  float* eb = e + (size_t)b * LL * LL;
  const int lm = lane & 15;
  const int hi = lane >> 4;        // lane half
  const int kA = hi * 8;           // 16-bit A frag: K = {kA..kA+7, kA+16..kA+23}
  const int kB = hi * 16;          // unused; B frag loads at kA, same K order as A

  v8f acc[4][4] = {};
  for (int dk = 0; dk < DD; dk += 32) {
    v16bf af[4], bf[4];
#pragma unroll
    for (int s = 0; s < 4; ++s) {
      const __bf16* pa = A + (size_t)(i0 + s * 16 + lm) * DD + dk + kA;
      v8bf alo = *(const v8bf*)pa;
      v8bf ahi = *(const v8bf*)(pa + 16);
      const __bf16* pb = Bm + (size_t)(j0 + s * 16 + lm) * DD + dk + kA;
      v8bf b0 = *(const v8bf*)pb;
      v8bf b1 = *(const v8bf*)(pb + 16);
#pragma unroll
      for (int k = 0; k < 8; ++k) {
        af[s][k] = alo[k]; af[s][k + 8] = ahi[k];
        bf[s][k] = b0[k];  bf[s][k + 8] = b1[k];
      }
      // unconditional global_prefetch_b8 of the next K slab (stays in d_ws)
      __builtin_prefetch(pa + 32, 0, 1);
      __builtin_prefetch(pb + 32, 0, 1);
    }
#pragma unroll
    for (int mi = 0; mi < 4; ++mi)
#pragma unroll
      for (int ni = 0; ni < 4; ++ni)
        acc[mi][ni] = __builtin_amdgcn_wmma_f32_16x16x32_bf16(
            false, af[mi], false, bf[ni], (short)0, acc[mi][ni], false, false);
  }
  // C/D layout: VGPR r -> M = r + 8*hi, lane%16 -> N
#pragma unroll
  for (int mi = 0; mi < 4; ++mi)
#pragma unroll
    for (int ni = 0; ni < 4; ++ni)
#pragma unroll
      for (int r = 0; r < 8; ++r)
        eb[(size_t)(i0 + mi * 16 + r + hi * 8) * LL + j0 + ni * 16 + lm] =
            acc[mi][ni][r];
}

// ---------------- row softmax stats: wave per row ----------------
__global__ __launch_bounds__(256) void k_rowstats(const float* __restrict__ e,
                                                  float* __restrict__ rowmax,
                                                  float* __restrict__ rinv) {
  const int lane = threadIdx.x & 31;
  const int row  = blockIdx.x * 8 + (threadIdx.x >> 5);   // over [0, 32*512)
  const float* p = e + (size_t)row * LL;
  float v[16];
#pragma unroll
  for (int k = 0; k < 16; ++k) v[k] = p[lane + 32 * k];
  float m = v[0];
#pragma unroll
  for (int k = 1; k < 16; ++k) m = fmaxf(m, v[k]);
#pragma unroll
  for (int off = 16; off > 0; off >>= 1) m = fmaxf(m, __shfl_xor(m, off, 32));
  float s = 0.f;
#pragma unroll
  for (int k = 0; k < 16; ++k) s += __expf(v[k] - m);
#pragma unroll
  for (int off = 16; off > 0; off >>= 1) s += __shfl_xor(s, off, 32);
  if (lane == 0) { rowmax[row] = m; rinv[row] = 1.0f / s; }
}

// ---------------- column softmax stats: thread per column ----------------
__global__ __launch_bounds__(256) void k_colstats(const float* __restrict__ e,
                                                  float* __restrict__ colmax,
                                                  float* __restrict__ cinv) {
  const int t = blockIdx.x * 256 + threadIdx.x;   // [0, 32*512)
  const int b = t >> 9;
  const int j = t & 511;
  const float* p = e + (size_t)b * LL * LL + j;
  float m = -3.4e38f;
  for (int i = 0; i < LL; ++i) m = fmaxf(m, p[(size_t)i * LL]);
  float s = 0.f;
  for (int i = 0; i < LL; ++i) s += __expf(p[(size_t)i * LL] - m);
  colmax[t] = m;
  cinv[t]   = 1.0f / s;
}

// ---------------- GEMM2: xe1 = e1 @ x2, xe2 = e2 @ x1 (fused) ----------------
// block = 256 thr = 8 waves as 2(i) x 4(d): block tile 32(i) x 256(d).
// B operands staged to LDS transposed ([d][j], stride 40 -> conflict-free
// ds_load_b128 reads: 16 lanes * 20-bank stride cover all 64 banks).
__global__ __launch_bounds__(256) void k_gemm2(const float* __restrict__ e,
                                               const __bf16* __restrict__ x1b,
                                               const __bf16* __restrict__ x2b,
                                               const float* __restrict__ rowmax,
                                               const float* __restrict__ rinv,
                                               const float* __restrict__ colmax,
                                               const float* __restrict__ cinv,
                                               float* __restrict__ out) {
  __shared__ __bf16 lds1[256 * 40];   // x1^T tile (for xe2)
  __shared__ __bf16 lds2[256 * 40];   // x2^T tile (for xe1)
  const int lane = threadIdx.x & 31;
  const int wave = threadIdx.x >> 5;
  const int b    = blockIdx.z;
  const int i0   = blockIdx.x * 32 + (wave >> 2) * 16;
  const int dblk = blockIdx.y * 256;
  const int dwav = (wave & 3) * 64;           // wave's d offset within block
  const int lm = lane & 15, hi = lane >> 4;
  const int kA = hi * 8;
  const int kB = hi * 16;
  const float*  eb  = e   + (size_t)b * LL * LL;
  const __bf16* X1  = x1b + (size_t)b * LL * DD;
  const __bf16* X2  = x2b + (size_t)b * LL * DD;
  const float*  cmb = colmax + b * LL;
  const float*  cib = cinv   + b * LL;
  const int   rowA = i0 + lm;
  const float rm   = rowmax[b * LL + rowA];
  // staging coords: lane-contiguous j, chunked d
  const int jj = threadIdx.x & 31;
  const int dc = (threadIdx.x >> 5) * 32;

  v8f acc1[4] = {}, acc2[4] = {};
  for (int j0 = 0; j0 < LL; j0 += 32) {
    __syncthreads();
    {
      const __bf16* s1 = X1 + (size_t)(j0 + jj) * DD + dblk + dc;
      const __bf16* s2 = X2 + (size_t)(j0 + jj) * DD + dblk + dc;
#pragma unroll
      for (int c = 0; c < 4; ++c) {
        v8bf a  = *(const v8bf*)(s1 + c * 8);
        v8bf bb = *(const v8bf*)(s2 + c * 8);
#pragma unroll
        for (int k = 0; k < 8; ++k) {
          lds1[(dc + c * 8 + k) * 40 + jj] = a[k];
          lds2[(dc + c * 8 + k) * 40 + jj] = bb[k];
        }
      }
    }
    __syncthreads();

    // A fragments: softmax weights regenerated from e tile
    const float* pe = eb + (size_t)rowA * LL + j0 + kA;
    f32x4 c0 = *(const f32x4*)pe;
    f32x4 c1 = *(const f32x4*)(pe + 4);
    f32x4 c2 = *(const f32x4*)(pe + 16);
    f32x4 c3 = *(const f32x4*)(pe + 20);
    f32x4 m0 = *(const f32x4*)(cmb + j0 + kA);
    f32x4 m1 = *(const f32x4*)(cmb + j0 + kA + 4);
    f32x4 m2 = *(const f32x4*)(cmb + j0 + kA + 16);
    f32x4 m3 = *(const f32x4*)(cmb + j0 + kA + 20);
    f32x4 s0 = *(const f32x4*)(cib + j0 + kA);
    f32x4 s1v = *(const f32x4*)(cib + j0 + kA + 4);
    f32x4 s2v = *(const f32x4*)(cib + j0 + kA + 16);
    f32x4 s3v = *(const f32x4*)(cib + j0 + kA + 20);
    v16bf a1, a2;
#pragma unroll
    for (int k = 0; k < 4; ++k) {
      a1[k]      = (__bf16)__expf(c0[k] - rm);
      a1[k + 4]  = (__bf16)__expf(c1[k] - rm);
      a1[k + 8]  = (__bf16)__expf(c2[k] - rm);
      a1[k + 12] = (__bf16)__expf(c3[k] - rm);
      a2[k]      = (__bf16)(__expf(c0[k] - m0[k]) * s0[k]);
      a2[k + 4]  = (__bf16)(__expf(c1[k] - m1[k]) * s1v[k]);
      a2[k + 8]  = (__bf16)(__expf(c2[k] - m2[k]) * s2v[k]);
      a2[k + 12] = (__bf16)(__expf(c3[k] - m3[k]) * s3v[k]);
    }

#pragma unroll
    for (int dt = 0; dt < 4; ++dt) {
      const int ldr = dwav + dt * 16 + lm;          // d index within block
      const __bf16* q1 = &lds1[ldr * 40 + kA];
      const __bf16* q2 = &lds2[ldr * 40 + kA];
      v8bf u0 = *(const v8bf*)q1;       v8bf u1 = *(const v8bf*)(q1 + 16);
      v8bf w0 = *(const v8bf*)q2;       v8bf w1 = *(const v8bf*)(q2 + 16);
      v16bf b1f, b2f;
#pragma unroll
      for (int k = 0; k < 8; ++k) {
        b1f[k] = u0[k]; b1f[k + 8] = u1[k];
        b2f[k] = w0[k]; b2f[k + 8] = w1[k];
      }
      acc1[dt] = __builtin_amdgcn_wmma_f32_16x16x32_bf16(
          false, a1, false, b2f, (short)0, acc1[dt], false, false);  // xe1: @x2
      acc2[dt] = __builtin_amdgcn_wmma_f32_16x16x32_bf16(
          false, a2, false, b1f, (short)0, acc2[dt], false, false);  // xe2: @x1
    }
  }

  // epilogue: fold 1/rowsum into xe1; write both outputs
  float rv[8];
  {
    f32x4 r0 = *(const f32x4*)(rinv + b * LL + i0 + hi * 8);
    f32x4 r1 = *(const f32x4*)(rinv + b * LL + i0 + hi * 8 + 4);
#pragma unroll
    for (int k = 0; k < 4; ++k) { rv[k] = r0[k]; rv[k + 4] = r1[k]; }
  }
  float* out1 = out;
  float* out2 = out + (size_t)NB * LL * DD;
#pragma unroll
  for (int dt = 0; dt < 4; ++dt)
#pragma unroll
    for (int r = 0; r < 8; ++r) {
      const int row = i0 + r + hi * 8;
      const int col = dblk + dwav + dt * 16 + lm;
      const size_t o = ((size_t)b * LL + row) * DD + col;
      out1[o] = acc1[dt][r] * rv[r];
      out2[o] = acc2[dt][r];
    }
}

// ---------------- launcher ----------------
extern "C" void kernel_launch(void* const* d_in, const int* in_sizes, int n_in,
                              void* d_out, int out_size, void* d_ws, size_t ws_size,
                              hipStream_t stream) {
  const float* x1 = (const float*)d_in[0];
  const float* x2 = (const float*)d_in[1];
  float* out = (float*)d_out;

  // scratch layout (bytes):
  //   [0,           25165824)  x1 bf16
  //   [25165824,    50331648)  x2 bf16
  //   [50331648,    83886080)  e fp32 [32,512,512]
  //   [83886080, ...)          rowmax | rinv | colmax | cinv (16384 f32 each)
  char* ws = (char*)d_ws;
  __bf16* x1b = (__bf16*)ws;
  __bf16* x2b = (__bf16*)(ws + 25165824u);
  float*  e   = (float*)(ws + 50331648u);
  float*  rowmax = (float*)(ws + 83886080u);
  float*  rinv   = rowmax + 16384;
  float*  colmax = rinv + 16384;
  float*  cinv   = colmax + 16384;

  const int n4 = NB * LL * DD / 4;   // 3145728
  k_cvt<<<(n4 + 255) / 256, 256, 0, stream>>>(x1, x1b, n4);
  k_cvt<<<(n4 + 255) / 256, 256, 0, stream>>>(x2, x2b, n4);
  k_gemm1<<<dim3(4, 2, NB), 256, 0, stream>>>(x1b, x2b, e);
  k_rowstats<<<(NB * LL) / 8, 256, 0, stream>>>(e, rowmax, rinv);
  k_colstats<<<(NB * LL) / 256, 256, 0, stream>>>(e, colmax, cinv);
  k_gemm2<<<dim3(LL / 32, DD / 256, NB), 256, 0, stream>>>(
      e, x1b, x2b, rowmax, rinv, colmax, cinv, out);
}
